// FP32GCN_23587960389847
// MI455X (gfx1250) — matrix operand order repaired and checked by reading, not compiled, not run on hardware
//
#include <hip/hip_runtime.h>
#include <hip/hip_bf16.h>

typedef __attribute__((ext_vector_type(16))) _Float16 v16h;
typedef __attribute__((ext_vector_type(8)))  _Float16 v8h;
typedef __attribute__((ext_vector_type(8)))  float    v8f;
typedef __attribute__((ext_vector_type(4)))  float    v4f;

#define HID      256
#define NNODES   1024
#define BATCH    256
#define M_TOTAL  (BATCH * NNODES)   // 262144 rows
#define KFEAT    131072             // N * HID/2

// ---------------------------------------------------------------------------
// Weight f32 -> f16 conversion (one-time, tiny)
// ---------------------------------------------------------------------------
__global__ void cvt_f32_f16(const float* __restrict__ in, _Float16* __restrict__ out, int n) {
    int i = blockIdx.x * 256 + threadIdx.x;
    if (i < n) out[i] = (_Float16)in[i];
}

// ---------------------------------------------------------------------------
// Layer 1 fused: image->graph + linear(K=3) + stencil aggregation + ReLU
// block = one (batch, node), 256 threads = output channels
// ---------------------------------------------------------------------------
__global__ __launch_bounds__(256) void gcn_layer1(
        const float* __restrict__ x, const float* __restrict__ W1,
        const float* __restrict__ b1, const float* __restrict__ adj,
        _Float16* __restrict__ out) {
    int bm = blockIdx.x;
    int b = bm >> 10, m = bm & 1023;
    int gi = m >> 5, gj = m & 31;
    __shared__ float xs[5][3];
    __shared__ float ac[5];
    __shared__ float s[4];
    int t = threadIdx.x;
    if (t < 5) {
        int n;
        if      (t == 0) n = m;
        else if (t == 1) n = (gi > 0)  ? m - 32 : -1;
        else if (t == 2) n = (gi < 31) ? m + 32 : -1;
        else if (t == 3) n = (gj > 0)  ? m - 1  : -1;
        else             n = (gj < 31) ? m + 1  : -1;
        float a = 0.f, x0 = 0.f, x1 = 0.f, x2 = 0.f;
        if (n >= 0) {
            a = adj[(size_t)m * 1024 + n];          // exact reference normalization
            const float* xb = x + (size_t)b * 3 * NNODES;
            x0 = xb[n]; x1 = xb[NNODES + n]; x2 = xb[2 * NNODES + n];
        }
        ac[t] = a; xs[t][0] = x0; xs[t][1] = x1; xs[t][2] = x2;
    }
    __syncthreads();
    if (t == 0) {
        float s0 = 0.f, s1 = 0.f, s2 = 0.f, sa = 0.f;
        #pragma unroll
        for (int j = 0; j < 5; j++) {
            s0 += ac[j] * xs[j][0]; s1 += ac[j] * xs[j][1];
            s2 += ac[j] * xs[j][2]; sa += ac[j];
        }
        s[0] = s0; s[1] = s1; s[2] = s2; s[3] = sa;
    }
    __syncthreads();
    // sum_j a_j*(x_j . w + b) = (sum a_j x_j) . w + b * sum a_j
    float v = s[0] * W1[t * 3] + s[1] * W1[t * 3 + 1] + s[2] * W1[t * 3 + 2] + s[3] * b1[t];
    out[(size_t)bm * HID + t] = (_Float16)fmaxf(v, 0.f);
}

// ---------------------------------------------------------------------------
// Stencil aggregation + ReLU on f16 activations: out[m,h] = relu(sum adj[m,n]*lin[n,h])
// ---------------------------------------------------------------------------
template <int H>
__global__ __launch_bounds__(H) void gcn_agg(
        const _Float16* __restrict__ lin, const float* __restrict__ adj,
        _Float16* __restrict__ out) {
    int bm = blockIdx.x;
    int b = bm >> 10, m = bm & 1023;
    int gi = m >> 5, gj = m & 31;
    __shared__ float ac[5];
    __shared__ int   nn[5];
    int t = threadIdx.x;
    if (t < 5) {
        int n;
        if      (t == 0) n = m;
        else if (t == 1) n = (gi > 0)  ? m - 32 : -1;
        else if (t == 2) n = (gi < 31) ? m + 32 : -1;
        else if (t == 3) n = (gj > 0)  ? m - 1  : -1;
        else             n = (gj < 31) ? m + 1  : -1;
        nn[t] = n;
        ac[t] = (n >= 0) ? adj[(size_t)m * 1024 + n] : 0.f;
    }
    __syncthreads();
    const _Float16* base = lin + (size_t)b * NNODES * H;
    float acc = 0.f;
    #pragma unroll
    for (int j = 0; j < 5; j++) {
        int n = nn[j];
        if (n >= 0) acc += ac[j] * (float)base[(size_t)n * H + t];
    }
    out[(size_t)bm * H + t] = (_Float16)fmaxf(acc, 0.f);
}

// ---------------------------------------------------------------------------
// WMMA GEMM: Out[M, N_OUT] = A[M, 256] * W[N_OUT, 256]^T + bias   (f16 in, f16 out, f32 acc)
// Block: 256 threads / 8 waves. Block tile 64(M) x N_OUT. Wave tile 16 x N_OUT/2.
// A tile staged to LDS (padded), fragments via ds_load_b128; B = row-major W rows
// (k order chosen here, same as A: each lane = one column, 16 contiguous K -> single 32B v16h load).
// ---------------------------------------------------------------------------
template <int N_OUT>
__global__ __launch_bounds__(256) void gcn_gemm(
        const _Float16* __restrict__ A, const _Float16* __restrict__ W,
        const float* __restrict__ bias, _Float16* __restrict__ Out) {
    constexpr int K    = 256;
    constexpr int BM   = 64;
    constexpr int NSUB = N_OUT / 32;           // 16-wide subtiles per wave
    __shared__ _Float16 As[BM][K + 8];         // row stride 528B: 16B aligned, bank-spread

    int t = threadIdx.x;
    size_t mBase = (size_t)blockIdx.x * BM;

    // ---- stage A tile (fully coalesced: 4 threads cover one row-chunk of 64B) ----
    {
        int row = t >> 2;                      // 0..63
        int kc  = (t & 3) * 8;                 // 0,8,16,24
        const _Float16* src = A + (mBase + row) * K + kc;
        #pragma unroll
        for (int c = 0; c < 8; c++)
            *(v8h*)&As[row][kc + c * 32] = *(const v8h*)(src + c * 32);
    }
    __syncthreads();

    int lane = t & 31, wave = t >> 5;
    int wm = wave >> 1;                        // 0..3  (M direction)
    int wn = wave & 1;                         // 0..1  (N direction)
    int llo = lane & 15, lhi = lane >> 4;
    int mrow  = wm * 16 + llo;                 // LDS row for A fragment
    int nbase = wn * (N_OUT / 2);

    v8f acc[NSUB];
    #pragma unroll
    for (int s = 0; s < NSUB; s++) acc[s] = v8f{0.f, 0.f, 0.f, 0.f, 0.f, 0.f, 0.f, 0.f};

    #pragma unroll
    for (int ks = 0; ks < K; ks += 32) {
        // A fragment: lanes 0-15 hold K {kb..kb+15}, lanes 16-31 shifted by 16
        int kb = ks + lhi * 16;
        v8h alo = *(const v8h*)&As[mrow][kb];
        v8h ahi = *(const v8h*)&As[mrow][kb + 8];
        v16h af;
        #pragma unroll
        for (int i = 0; i < 8; i++) { af[i] = alo[i]; af[8 + i] = ahi[i]; }

        // B fragments: lane holds column n, 16 contiguous K values
        int kcol = ks + lhi * 16;
        #pragma unroll
        for (int s = 0; s < NSUB; s++) {
            int n = nbase + s * 16 + llo;
            v16h bf = *(const v16h*)(W + (size_t)n * K + kcol);
            acc[s] = __builtin_amdgcn_wmma_f32_16x16x32_f16(
                false, af, false, bf, (short)0, acc[s], false, false);
        }
    }

    // ---- epilogue: bias add, f16 store (coalesced 32B segments per half-wave per VGPR) ----
    #pragma unroll
    for (int s = 0; s < NSUB; s++) {
        int n = nbase + s * 16 + llo;
        float bn = bias[n];
        size_t rbase = mBase + wm * 16 + lhi * 8;
        #pragma unroll
        for (int v = 0; v < 8; v++)
            Out[(rbase + v) * N_OUT + n] = (_Float16)(acc[s][v] + bn);
    }
}

// ---------------------------------------------------------------------------
// Classifier: out[b,c] = sum_k h[b,k]*Wc[c,k] + bc[c]   (K = 131072, 10 classes)
// One block per batch element; wave32 shuffle reduce + LDS cross-wave combine.
// ---------------------------------------------------------------------------
__global__ __launch_bounds__(256) void gcn_classifier(
        const _Float16* __restrict__ h, const float* __restrict__ Wc,
        const float* __restrict__ bc, float* __restrict__ out) {
    int b = blockIdx.x, t = threadIdx.x;
    const _Float16* hb = h + (size_t)b * KFEAT;
    float acc[10];
    #pragma unroll
    for (int c = 0; c < 10; c++) acc[c] = 0.f;

    for (int base = t * 8; base < KFEAT; base += 2048) {
        v8h hv = *(const v8h*)(hb + base);
        float hf[8];
        #pragma unroll
        for (int i = 0; i < 8; i++) hf[i] = (float)hv[i];
        #pragma unroll
        for (int c = 0; c < 10; c++) {
            const float* w = Wc + (size_t)c * KFEAT + base;
            v4f w0 = *(const v4f*)w;
            v4f w1 = *(const v4f*)(w + 4);
            acc[c] += hf[0]*w0[0] + hf[1]*w0[1] + hf[2]*w0[2] + hf[3]*w0[3]
                    + hf[4]*w1[0] + hf[5]*w1[1] + hf[6]*w1[2] + hf[7]*w1[3];
        }
    }

    __shared__ float red[8][10];
    int lane = t & 31, wave = t >> 5;
    #pragma unroll
    for (int c = 0; c < 10; c++) {
        float r = acc[c];
        #pragma unroll
        for (int off = 16; off > 0; off >>= 1) r += __shfl_down(r, off, 32);
        if (lane == 0) red[wave][c] = r;
    }
    __syncthreads();
    if (t < 10) {
        float r = bc[t];
        #pragma unroll
        for (int w = 0; w < 8; w++) r += red[w][t];
        out[b * 10 + t] = r;
    }
}

// ---------------------------------------------------------------------------
extern "C" void kernel_launch(void* const* d_in, const int* in_sizes, int n_in,
                              void* d_out, int out_size, void* d_ws, size_t ws_size,
                              hipStream_t stream) {
    (void)in_sizes; (void)n_in; (void)out_size; (void)ws_size;
    const float* x   = (const float*)d_in[0];
    const float* W1  = (const float*)d_in[1];
    const float* b1  = (const float*)d_in[2];
    const float* W2  = (const float*)d_in[3];
    const float* b2  = (const float*)d_in[4];
    const float* W3  = (const float*)d_in[5];
    const float* b3  = (const float*)d_in[6];
    const float* Wc  = (const float*)d_in[7];
    const float* bc  = (const float*)d_in[8];
    const float* adj = (const float*)d_in[9];

    char* ws = (char*)d_ws;
    _Float16* w2h  = (_Float16*)ws;                                   // 256*256 halves
    _Float16* w3h  = (_Float16*)(ws + 131072);                        // 128*256 halves
    _Float16* bufA = (_Float16*)(ws + 196608);                        // 262144*256 halves (128MB)
    _Float16* bufB = (_Float16*)(ws + 196608 + (size_t)M_TOTAL * HID * 2);

    cvt_f32_f16<<<(HID * HID + 255) / 256, 256, 0, stream>>>(W2, w2h, HID * HID);
    cvt_f32_f16<<<(128 * HID + 255) / 256, 256, 0, stream>>>(W3, w3h, 128 * HID);

    // L1: fused linear(K=3) + stencil + relu -> bufA [M,256] f16
    gcn_layer1<<<M_TOTAL, 256, 0, stream>>>(x, W1, b1, adj, bufA);
    // L2: WMMA GEMM -> bufB, stencil+relu -> bufA
    gcn_gemm<256><<<M_TOTAL / 64, 256, 0, stream>>>(bufA, w2h, b2, bufB);
    gcn_agg<256><<<M_TOTAL, 256, 0, stream>>>(bufB, adj, bufA);
    // L3: WMMA GEMM (N=128) -> bufB, stencil+relu -> bufA [M,128]
    gcn_gemm<128><<<M_TOTAL / 64, 256, 0, stream>>>(bufA, w3h, b3, bufB);
    gcn_agg<128><<<M_TOTAL, 128, 0, stream>>>(bufB, adj, bufA);
    // Classifier -> d_out [256,10] f32
    gcn_classifier<<<BATCH, 256, 0, stream>>>(bufA, Wc, bc, (float*)d_out);
}
